// Apriel2GatedDeltaNet_45621142618777
// MI455X (gfx1250) — compile-verified, matrix operand order checked by reading, not run on hardware
//
#include <hip/hip_runtime.h>
#include <hip/hip_bf16.h>
#include <math.h>

// ---------------- problem constants ----------------
#define B   2
#define T   1024
#define H   4096
#define HK  8
#define HV  32
#define DK  64
#define DV  64
#define VPK 4                    // HV / HK
#define KEY_DIM  512             // HK*DK
#define VAL_DIM  2048            // HV*DV
#define CONV_DIM 3072            // 2*KEY_DIM + VAL_DIM
#define QKVZ_N   5120            // 2*KEY_DIM + 2*VAL_DIM
#define KSZ 4

typedef __bf16 bf16;
typedef __attribute__((ext_vector_type(16))) __bf16 v16bf;
typedef __attribute__((ext_vector_type(8)))  __bf16 v8bf;
typedef __attribute__((ext_vector_type(8)))  float  v8f;

// ---------------- fp32 -> bf16 convert (row-major, no transpose) ----------------
__global__ __launch_bounds__(256) void cvt_bf16_kernel(const float* __restrict__ in,
                                                       bf16* __restrict__ out, size_t n) {
    size_t i = (size_t)blockIdx.x * blockDim.x + threadIdx.x;
    if (i < n) out[i] = (bf16)in[i];
}

// ---------------- fp32 (K x N) -> bf16 transposed (N x K), LDS tiled ----------------
__global__ __launch_bounds__(256) void transpose_cvt_kernel(const float* __restrict__ in,
                                                            bf16* __restrict__ out,
                                                            int K, int N) {
    __shared__ bf16 tile[32][33];
    int kb = blockIdx.y * 32, nb = blockIdx.x * 32;
    int tx = threadIdx.x & 31, ty = threadIdx.x >> 5;   // 32x8 = 256 threads
    #pragma unroll
    for (int r = ty; r < 32; r += 8) {
        int k = kb + r, n = nb + tx;
        tile[r][tx] = (k < K && n < N) ? (bf16)in[(size_t)k * N + n] : (bf16)0.0f;
    }
    __syncthreads();
    #pragma unroll
    for (int r = ty; r < 32; r += 8) {
        int n = nb + r, k = kb + tx;
        if (n < N && k < K) out[(size_t)n * K + k] = tile[tx][r];
    }
}

// ---------------- bf16 WMMA GEMM: C[M,N](f32) = A[M,K](bf16) * Bt[N,K](bf16) ----------------
// one wave per 64x32 tile (4 M-frags x 2 N-frags); __launch_bounds__(256,1) so the
// allocator keeps 8 f32 accumulators + 6 bf16 fragments fully in VGPRs (no spills).
__global__ __launch_bounds__(256, 1)
void gemm_bf16_wmma_kernel(const bf16* __restrict__ A, const bf16* __restrict__ Bt,
                           float* __restrict__ C, int M, int N, int K) {
    const int tiles_n = N >> 5;                       // 32-wide N tiles
    const int total   = (M >> 6) * tiles_n;           // 64-tall M tiles
    const int wave    = threadIdx.x >> 5;
    const int tile    = blockIdx.x * (blockDim.x >> 5) + wave;
    if (tile >= total) return;                        // wave-uniform: EXEC stays all-ones
    const int tn = tile % tiles_n, tm = tile / tiles_n;
    const int lane = threadIdx.x & 31;
    const int half = (lane >= 16) ? 1 : 0;            // K-half select per ISA layout
    const int l16  = lane & 15;
    const int m0 = tm << 6, n0 = tn << 5;

    // per-lane fragment pointers, advanced by 32 elements per K-step
    const bf16* ap0 = A + (size_t)(m0 +  0 + l16) * K + half * 8;
    const bf16* ap1 = A + (size_t)(m0 + 16 + l16) * K + half * 8;
    const bf16* ap2 = A + (size_t)(m0 + 32 + l16) * K + half * 8;
    const bf16* ap3 = A + (size_t)(m0 + 48 + l16) * K + half * 8;
    const bf16* bp0 = Bt + (size_t)(n0 +  0 + l16) * K + half * 8;
    const bf16* bp1 = Bt + (size_t)(n0 + 16 + l16) * K + half * 8;

    v8f acc[4][2];
    #pragma unroll
    for (int i = 0; i < 4; i++)
        #pragma unroll
        for (int j = 0; j < 2; j++)
            #pragma unroll
            for (int r = 0; r < 8; r++) acc[i][j][r] = 0.0f;

    for (int k0 = 0; k0 < K; k0 += 32) {
        // fragment = lanes' K in [kb..kb+7] (low b128) and [kb+16..kb+23] (high b128)
        v8bf a0l = *(const v8bf*)ap0, a0h = *(const v8bf*)(ap0 + 16);
        v8bf a1l = *(const v8bf*)ap1, a1h = *(const v8bf*)(ap1 + 16);
        v8bf a2l = *(const v8bf*)ap2, a2h = *(const v8bf*)(ap2 + 16);
        v8bf a3l = *(const v8bf*)ap3, a3h = *(const v8bf*)(ap3 + 16);
        v8bf b0l = *(const v8bf*)bp0, b0h = *(const v8bf*)(bp0 + 16);
        v8bf b1l = *(const v8bf*)bp1, b1h = *(const v8bf*)(bp1 + 16);
        ap0 += 32; ap1 += 32; ap2 += 32; ap3 += 32; bp0 += 32; bp1 += 32;

        v16bf af[4], bfr[2];
        af[0]  = __builtin_shufflevector(a0l, a0h, 0,1,2,3,4,5,6,7,8,9,10,11,12,13,14,15);
        af[1]  = __builtin_shufflevector(a1l, a1h, 0,1,2,3,4,5,6,7,8,9,10,11,12,13,14,15);
        af[2]  = __builtin_shufflevector(a2l, a2h, 0,1,2,3,4,5,6,7,8,9,10,11,12,13,14,15);
        af[3]  = __builtin_shufflevector(a3l, a3h, 0,1,2,3,4,5,6,7,8,9,10,11,12,13,14,15);
        bfr[0] = __builtin_shufflevector(b0l, b0h, 0,1,2,3,4,5,6,7,8,9,10,11,12,13,14,15);
        bfr[1] = __builtin_shufflevector(b1l, b1h, 0,1,2,3,4,5,6,7,8,9,10,11,12,13,14,15);

        #pragma unroll
        for (int i = 0; i < 4; i++)
            #pragma unroll
            for (int j = 0; j < 2; j++)
                acc[i][j] = __builtin_amdgcn_wmma_f32_16x16x32_bf16(
                    false, af[i], false, bfr[j], (short)0, acc[i][j], false, false);
    }
    // C/D layout: lane -> col n0 + j*16 + l16 ; vgpr r -> row m + r + 8*half
    #pragma unroll
    for (int i = 0; i < 4; i++)
        #pragma unroll
        for (int j = 0; j < 2; j++)
            #pragma unroll
            for (int r = 0; r < 8; r++)
                C[(size_t)(m0 + i * 16 + r + half * 8) * N + (n0 + j * 16 + l16)] = acc[i][j][r];
}

// ---------------- fused conv(KSZ=4)+SiLU, L2-norms, gates ----------------
// one block (256 thr) per (b,t)
__global__ __launch_bounds__(256)
void prep_kernel(const float* __restrict__ qkvz, const float* __restrict__ ba,
                 const float* __restrict__ conv_w, const float* __restrict__ dt_bias,
                 const float* __restrict__ A_log,
                 float* __restrict__ qn, float* __restrict__ kn, float* __restrict__ vv,
                 float* __restrict__ gg, float* __restrict__ bb) {
    const int bt = blockIdx.x;                 // b*T + t
    const int b = bt / T, t = bt % T;
    __shared__ float conv_s[CONV_DIM];
    const int tid = threadIdx.x;

    for (int c = tid; c < CONV_DIM; c += 256) {
        int off;
        if (c < KEY_DIM)            { int hk = c >> 6,          d = c & 63;           off = hk * 640 + d; }
        else if (c < 2 * KEY_DIM)   { int c2 = c - KEY_DIM;     int hk = c2 >> 6;     off = hk * 640 + 64 + (c2 & 63); }
        else                        { int c3 = c - 2 * KEY_DIM; int hk = c3 >> 8;     off = hk * 640 + 128 + (c3 & 255); }
        float s = 0.0f;
        #pragma unroll
        for (int j = 0; j < KSZ; j++) {
            int tt = t - (KSZ - 1) + j;
            float x = (tt >= 0) ? qkvz[((size_t)b * T + tt) * QKVZ_N + off] : 0.0f;
            s = fmaf(x, conv_w[c * KSZ + j], s);
        }
        conv_s[c] = s / (1.0f + __expf(-s));   // SiLU
    }
    __syncthreads();

    const int wave = tid >> 5, lane = tid & 31;
    if (wave < HK) {                            // wave w: q-head w and k-head w (64 dims each)
        float x0 = conv_s[wave * 64 + lane], x1 = conv_s[wave * 64 + lane + 32];
        float ss = x0 * x0 + x1 * x1;
        #pragma unroll
        for (int m = 16; m >= 1; m >>= 1) ss += __shfl_xor(ss, m, 32);
        float invq = rsqrtf(ss + 1e-6f) * 0.125f;   // * DK^-0.5
        size_t base = ((size_t)bt * HK + wave) * 64;
        qn[base + lane] = x0 * invq;  qn[base + lane + 32] = x1 * invq;

        float y0 = conv_s[KEY_DIM + wave * 64 + lane], y1 = conv_s[KEY_DIM + wave * 64 + lane + 32];
        float tt2 = y0 * y0 + y1 * y1;
        #pragma unroll
        for (int m = 16; m >= 1; m >>= 1) tt2 += __shfl_xor(tt2, m, 32);
        float invk = rsqrtf(tt2 + 1e-6f);
        kn[base + lane] = y0 * invk;  kn[base + lane + 32] = y1 * invk;
    }
    for (int c = tid; c < VAL_DIM; c += 256)
        vv[(size_t)bt * VAL_DIM + c] = conv_s[2 * KEY_DIM + c];

    if (tid < HV) {                             // gates: ba is (B,T,HK,2*VPK)
        int h = tid, hk = h >> 2, r = h & 3;
        float bval = ba[(size_t)bt * (2 * HV) + hk * 8 + r];
        float aval = ba[(size_t)bt * (2 * HV) + hk * 8 + VPK + r];
        float x  = aval + dt_bias[h];
        float sp = (x > 20.0f) ? x : log1pf(__expf(x));      // softplus
        gg[(size_t)bt * HV + h] = -__expf(A_log[h]) * sp;
        bb[(size_t)bt * HV + h] = 1.0f / (1.0f + __expf(-bval));
    }
}

// ---------------- sequential delta-rule scan: 64 blocks (b,h), 64 thr = state columns ----------------
#define TC 8
__global__ __launch_bounds__(64, 1)
void scan_kernel(const float* __restrict__ qn, const float* __restrict__ kn,
                 const float* __restrict__ vv, const float* __restrict__ gg,
                 const float* __restrict__ bb, float* __restrict__ o) {
    const int bh = blockIdx.x;                 // 0..B*HV-1
    const int b = bh >> 5, h = bh & 31, hk = h >> 2;
    const int tid = threadIdx.x;               // 0..63 : v-column of state
    float S[DK];
    #pragma unroll
    for (int i = 0; i < DK; i++) S[i] = 0.0f;

    __shared__ float qs[TC * DK], ks[TC * DK], vs[TC * DV], gs[TC], bs[TC];

    for (int t0 = 0; t0 < T; t0 += TC) {
        for (int i = tid; i < TC * 64; i += 64) {
            int tt = t0 + (i >> 6), d = i & 63;
            size_t qk = ((size_t)(b * T + tt) * HK + hk) * DK + d;
            qs[i] = qn[qk];
            ks[i] = kn[qk];
            vs[i] = vv[((size_t)(b * T + tt) * HV + h) * DV + d];
        }
        if (tid < TC) {
            gs[tid] = gg[(size_t)(b * T + t0 + tid) * HV + h];
            bs[tid] = bb[(size_t)(b * T + t0 + tid) * HV + h];
        }
        __syncthreads();
        for (int s = 0; s < TC; s++) {
            const float eg = __expf(gs[s]);
            const float* kk = ks + s * DK;
            const float* qq = qs + s * DK;
            float pred = 0.0f;
            #pragma unroll
            for (int i = 0; i < DK; i++) { S[i] *= eg; pred = fmaf(kk[i], S[i], pred); }
            const float delta = (vs[s * DV + tid] - pred) * bs[s];
            float oacc = 0.0f;
            #pragma unroll
            for (int i = 0; i < DK; i++) { S[i] = fmaf(kk[i], delta, S[i]); oacc = fmaf(qq[i], S[i], oacc); }
            o[((size_t)(b * T + t0 + s) * HV + h) * DV + tid] = oacc;
        }
        __syncthreads();
    }
}

// ---------------- RMS-norm * silu(z) gating -> bf16 for final GEMM ----------------
__global__ __launch_bounds__(256)
void gate_kernel(const float* __restrict__ o, const float* __restrict__ qkvz,
                 const float* __restrict__ norm_w, bf16* __restrict__ og) {
    const int gid  = blockIdx.x * (blockDim.x >> 5) + (threadIdx.x >> 5); // (bt,h) per wave
    const int lane = threadIdx.x & 31;
    if (gid >= B * T * HV) return;
    const int h = gid & 31, bt = gid >> 5;
    const int hk = h >> 2, r = h & 3;
    const float* orow = o + (size_t)gid * DV;
    float x0 = orow[lane], x1 = orow[lane + 32];
    float ss = x0 * x0 + x1 * x1;
    #pragma unroll
    for (int m = 16; m >= 1; m >>= 1) ss += __shfl_xor(ss, m, 32);
    const float inv = rsqrtf(ss * (1.0f / DV) + 1e-5f);
    const size_t zoff = (size_t)bt * QKVZ_N + hk * 640 + 128 + VPK * DV + r * DV;
    float z0 = qkvz[zoff + lane], z1 = qkvz[zoff + lane + 32];
    float g0 = x0 * inv * norm_w[lane]      * (z0 / (1.0f + __expf(-z0)));
    float g1 = x1 * inv * norm_w[lane + 32] * (z1 / (1.0f + __expf(-z1)));
    og[(size_t)gid * DV + lane]      = (bf16)g0;
    og[(size_t)gid * DV + lane + 32] = (bf16)g1;
}

// ---------------- host-side orchestration ----------------
extern "C" void kernel_launch(void* const* d_in, const int* in_sizes, int n_in,
                              void* d_out, int out_size, void* d_ws, size_t ws_size,
                              hipStream_t stream) {
    (void)in_sizes; (void)n_in; (void)out_size; (void)ws_size;
    const float* hs      = (const float*)d_in[0];
    const float* W_qkvz  = (const float*)d_in[1];
    const float* W_ba    = (const float*)d_in[2];
    const float* conv_w  = (const float*)d_in[3];
    const float* dt_bias = (const float*)d_in[4];
    const float* A_log   = (const float*)d_in[5];
    const float* norm_w  = (const float*)d_in[6];
    const float* W_out   = (const float*)d_in[7];
    float* out = (float*)d_out;

    const int BT = B * T;                       // 2048
    char* w = (char*)d_ws;
    auto alloc = [&](size_t bytes) { void* p = (void*)w; w += (bytes + 255) & ~(size_t)255; return p; };

    bf16*  hsb    = (bf16*)alloc((size_t)BT * H * 2);
    bf16*  wqkT   = (bf16*)alloc((size_t)QKVZ_N * H * 2);
    bf16*  wbaT   = (bf16*)alloc((size_t)(2 * HV) * H * 2);
    bf16*  woutT  = (bf16*)alloc((size_t)H * VAL_DIM * 2);
    float* qkvz   = (float*)alloc((size_t)BT * QKVZ_N * 4);
    float* ba     = (float*)alloc((size_t)BT * 2 * HV * 4);
    float* qn     = (float*)alloc((size_t)BT * KEY_DIM * 4);
    float* kn     = (float*)alloc((size_t)BT * KEY_DIM * 4);
    float* vv     = (float*)alloc((size_t)BT * VAL_DIM * 4);
    float* gg     = (float*)alloc((size_t)BT * HV * 4);
    float* bb     = (float*)alloc((size_t)BT * HV * 4);
    float* o      = (float*)alloc((size_t)BT * VAL_DIM * 4);
    bf16*  og     = (bf16*)alloc((size_t)BT * VAL_DIM * 2);

    // 1) activations -> bf16
    {
        size_t n = (size_t)BT * H;
        cvt_bf16_kernel<<<(unsigned)((n + 255) / 256), 256, 0, stream>>>(hs, hsb, n);
    }
    // 2) weights -> bf16 transposed (K-contiguous for WMMA fragments)
    transpose_cvt_kernel<<<dim3(QKVZ_N / 32, H / 32), 256, 0, stream>>>(W_qkvz, wqkT, H, QKVZ_N);
    transpose_cvt_kernel<<<dim3((2 * HV) / 32, H / 32), 256, 0, stream>>>(W_ba, wbaT, H, 2 * HV);
    transpose_cvt_kernel<<<dim3(H / 32, VAL_DIM / 32), 256, 0, stream>>>(W_out, woutT, VAL_DIM, H);

    // 3) qkvz = hs @ W_qkvz   (2048 x 5120, K=4096) : 32*160=5120 tiles, 8 waves/block
    gemm_bf16_wmma_kernel<<<(BT / 64) * (QKVZ_N / 32) / 8, 256, 0, stream>>>(hsb, wqkT, qkvz, BT, QKVZ_N, H);
    // 4) ba = hs @ W_ba       (2048 x 64, K=4096)   : 64 tiles
    gemm_bf16_wmma_kernel<<<(BT / 64) * ((2 * HV) / 32) / 8, 256, 0, stream>>>(hsb, wbaT, ba, BT, 2 * HV, H);

    // 5) conv + SiLU + L2 norms + gates
    prep_kernel<<<BT, 256, 0, stream>>>(qkvz, ba, conv_w, dt_bias, A_log, qn, kn, vv, gg, bb);

    // 6) sequential gated delta-rule scan
    scan_kernel<<<B * HV, 64, 0, stream>>>(qn, kn, vv, gg, bb, o);

    // 7) RMS norm + silu(z) gate -> bf16
    gate_kernel<<<(BT * HV) / 8, 256, 0, stream>>>(o, qkvz, norm_w, og);

    // 8) out = og @ W_out     (2048 x 4096, K=2048) : 32*128=4096 tiles
    gemm_bf16_wmma_kernel<<<(BT / 64) * (H / 32) / 8, 256, 0, stream>>>(og, woutT, out, BT, H, VAL_DIM);
}
